// PatchedDbrxAttention_29162827940484
// MI455X (gfx1250) — matrix operand order repaired and checked by reading, not compiled, not run on hardware
//
#include <hip/hip_runtime.h>
#include <hip/hip_bf16.h>

// ---------------------------------------------------------------------------
// Problem constants (from the reference)
// ---------------------------------------------------------------------------
#define D_MODEL   4096
#define N_HEADS   32
#define N_KV      8
#define GQA_G     4            // N_HEADS / N_KV
#define HEAD_DIM  128
#define BATCH     4
#define Q_LEN     1024
#define TOTAL     (BATCH * Q_LEN)          // 4096 tokens
#define QKV_DIM   ((N_HEADS + 2 * N_KV) * HEAD_DIM)   // 6144
#define Q_COLS    (N_HEADS * HEAD_DIM)     // 4096
#define K_COL0    Q_COLS                   // 4096
#define V_COL0    (Q_COLS + N_KV * HEAD_DIM) // 5120
#define CLIP_QKV  8.0f

// ---------------------------------------------------------------------------
// CDNA5 WMMA types: v_wmma_f32_16x16x32_bf16, D(16x16 f32) = A(16x32) B(32x16)
// ---------------------------------------------------------------------------
typedef __attribute__((ext_vector_type(16))) __bf16 v16bf;
typedef __attribute__((ext_vector_type(8)))  float  v8f;

union Frag {               // 32 bytes: one WMMA A/B operand per lane
    v16bf          v;
    uint4          q[2];
    unsigned short s[16];
};

__device__ __forceinline__ unsigned short f2bf(float x) {
    union { float f; unsigned u; } c; c.f = x;
    unsigned u = c.u + 0x7FFFu + ((c.u >> 16) & 1u);   // round-to-nearest-even
    return (unsigned short)(u >> 16);
}

// ---------------------------------------------------------------------------
// Kernel 1/4: generic  C[M,N] = A[M,K] * W[N,K]^T   (fp32 in/out, bf16 WMMA)
// Tile: 128x128x32 per 256-thread block; 8 waves as 2(M) x 4(N), each wave
// owns a 64x32 sub-tile = 4x2 WMMA accumulators.
// ---------------------------------------------------------------------------
__global__ __launch_bounds__(256) void gemm_bf16_tn(
    const float* __restrict__ A,    // [M,K] row-major
    const float* __restrict__ W,    // [N,K] row-major (weight rows = out cols)
    float*       __restrict__ C,    // [M,N]
    int M, int N, int K)
{
    __shared__ unsigned short ldsA[128 * 32];   // [m][k] bf16
    __shared__ unsigned short ldsB[128 * 32];   // [n][k] bf16

    const int tid  = threadIdx.x;
    const int lane = tid & 31;
    const int w    = tid >> 5;        // 0..7
    const int wm   = w >> 2;          // 0..1 -> 64-row half
    const int wn   = w & 3;           // 0..3 -> 32-col strip
    const int l16  = lane & 15;
    const int h16  = lane >> 4;
    const int m0   = blockIdx.y * 128;
    const int n0   = blockIdx.x * 128;

    v8f acc[4][2];
    for (int i = 0; i < 4; ++i)
        for (int j = 0; j < 2; ++j)
            acc[i][j] = (v8f)0.0f;

    const int srow  = tid >> 1;          // 0..127
    const int shalf = (tid & 1) * 16;    // 0 or 16 (k offset)

    for (int kb = 0; kb < K; kb += 32) {
        // ---- stage fp32 -> bf16 LDS tiles (each thread: 16 floats of A, 16 of W)
        const float* ga = A + (size_t)(m0 + srow) * K + kb + shalf;
        const float* gb = W + (size_t)(n0 + srow) * K + kb + shalf;
        if (kb + 32 < K) {                      // global_prefetch_b8 next K tile
            __builtin_prefetch(ga + 32, 0, 1);
            __builtin_prefetch(gb + 32, 0, 1);
        }
        unsigned short* da = &ldsA[srow * 32 + shalf];
        unsigned short* db = &ldsB[srow * 32 + shalf];
        #pragma unroll
        for (int i = 0; i < 4; ++i) {
            float4 av = ((const float4*)ga)[i];
            float4 bv = ((const float4*)gb)[i];
            da[i * 4 + 0] = f2bf(av.x); da[i * 4 + 1] = f2bf(av.y);
            da[i * 4 + 2] = f2bf(av.z); da[i * 4 + 3] = f2bf(av.w);
            db[i * 4 + 0] = f2bf(bv.x); db[i * 4 + 1] = f2bf(bv.y);
            db[i * 4 + 2] = f2bf(bv.z); db[i * 4 + 3] = f2bf(bv.w);
        }
        __syncthreads();

        // ---- A fragments: 16x32, lane row = l16; K chunks {h16*8..+8, 16+h16*8..+8}
        Frag af[4];
        #pragma unroll
        for (int ms = 0; ms < 4; ++ms) {
            int off = (wm * 64 + ms * 16 + l16) * 32 + h16 * 8;
            af[ms].q[0] = *(const uint4*)&ldsA[off];
            af[ms].q[1] = *(const uint4*)&ldsA[off + 16];
        }
        // ---- B fragments: 32x16, lane col = l16; K chunks {h16*8..+8, 16+h16*8..+8}
        Frag bf[2];
        #pragma unroll
        for (int ns = 0; ns < 2; ++ns) {
            int off = (wn * 32 + ns * 16 + l16) * 32 + h16 * 8;
            bf[ns].q[0] = *(const uint4*)&ldsB[off];
            bf[ns].q[1] = *(const uint4*)&ldsB[off + 16];
        }
        #pragma unroll
        for (int ms = 0; ms < 4; ++ms)
            #pragma unroll
            for (int ns = 0; ns < 2; ++ns)
                acc[ms][ns] = __builtin_amdgcn_wmma_f32_16x16x32_bf16(
                    false, af[ms].v, false, bf[ns].v,
                    (short)0, acc[ms][ns], false, false);
        __syncthreads();
    }

    // ---- epilogue: C layout M = r + 8*h16, N = l16
    #pragma unroll
    for (int ms = 0; ms < 4; ++ms)
        #pragma unroll
        for (int ns = 0; ns < 2; ++ns) {
            int row = m0 + wm * 64 + ms * 16 + 8 * h16;
            int col = n0 + wn * 32 + ns * 16 + l16;
            float* cp = C + (size_t)row * N + col;
            #pragma unroll
            for (int r = 0; r < 8; ++r)
                cp[(size_t)r * N] = acc[ms][ns][r];
        }
}

// ---------------------------------------------------------------------------
// Kernel 2/4: in-place clamp(+-8) and NeoX rotate-half RoPE on q,k (clamp only
// on v). One thread per (token, head, dim-pair).
// ---------------------------------------------------------------------------
#define ROPE_HEADS (N_HEADS + 2 * N_KV)     // 48
__global__ void rope_clamp_kernel(float* __restrict__ qkv,
                                  const float* __restrict__ inv_freq,
                                  const int*   __restrict__ pos_ids)
{
    const int PAIRS = ROPE_HEADS * (HEAD_DIM / 2);     // 48 * 64
    int idx = blockIdx.x * blockDim.x + threadIdx.x;
    if (idx >= TOTAL * PAIRS) return;
    int t  = idx / PAIRS;
    int r  = idx - t * PAIRS;
    int hh = r >> 6;           // 0..47
    int d  = r & 63;           // dim-pair index

    int  col;
    bool dorope;
    if (hh < N_HEADS)            { col = hh * HEAD_DIM;                        dorope = true;  }
    else if (hh < N_HEADS + N_KV){ col = K_COL0 + (hh - N_HEADS) * HEAD_DIM;   dorope = true;  }
    else                         { col = V_COL0 + (hh - N_HEADS - N_KV) * HEAD_DIM; dorope = false; }

    float* row = qkv + (size_t)t * QKV_DIM;
    float x1 = row[col + d];
    float x2 = row[col + 64 + d];
    x1 = fminf(fmaxf(x1, -CLIP_QKV), CLIP_QKV);
    x2 = fminf(fmaxf(x2, -CLIP_QKV), CLIP_QKV);
    if (dorope) {
        float f = (float)pos_ids[t] * inv_freq[d];
        float s, c;
        __sincosf(f, &s, &c);
        float o1 = x1 * c - x2 * s;
        float o2 = x2 * c + x1 * s;
        x1 = o1; x2 = o2;
    }
    row[col + d]      = x1;
    row[col + 64 + d] = x2;
}

// ---------------------------------------------------------------------------
// Kernel 3/4: causal GQA flash attention, bf16 WMMA.
// Block = 128 threads (4 waves) = one (batch, kv-head, 16-row Q tile).
// Wave w handles query head hk*4+w. K/V^T chunks (32 rows) staged in LDS and
// shared by the 4 heads. Softmax statistics live in the WMMA C layout
// (row = r + 8*(lane>>4), col = lane&15) with width-16 shuffle reductions.
// ---------------------------------------------------------------------------
__global__ __launch_bounds__(128) void attn_kernel(
    const float* __restrict__ qkv,     // roped/clamped [TOTAL, 6144]
    float*       __restrict__ attn)    // [TOTAL, 4096]
{
    __shared__ unsigned short ldsK [32 * 128];      // [kv][d]   bf16
    __shared__ unsigned short ldsVt[128 * 32];      // [d][kv]   bf16
    __shared__ unsigned short ldsP [GQA_G][16 * 32];// per-wave P: [q][kv] bf16

    const int lane = threadIdx.x & 31;
    const int w    = threadIdx.x >> 5;   // head within GQA group
    const int l16  = lane & 15;
    const int h16  = lane >> 4;
    const int q0   = blockIdx.x * 16;
    const int hk   = blockIdx.y;
    const int b    = blockIdx.z;
    const int h    = hk * GQA_G + w;
    const float scale = 0.08838834764831845f;   // 1/sqrt(128)

    // ---- Q fragments (A layout): row = l16, K chunks {h8..h8+8, 16+h8..}
    Frag qf[4];
    {
        const float* qrow = qkv + (size_t)(b * Q_LEN + q0 + l16) * QKV_DIM + h * HEAD_DIM;
        #pragma unroll
        for (int kf = 0; kf < 4; ++kf) {
            const float* p0 = qrow + kf * 32 + h16 * 8;
            const float* p1 = p0 + 16;
            float4 a0 = ((const float4*)p0)[0], a1 = ((const float4*)p0)[1];
            float4 b0 = ((const float4*)p1)[0], b1 = ((const float4*)p1)[1];
            qf[kf].s[0]  = f2bf(a0.x); qf[kf].s[1]  = f2bf(a0.y);
            qf[kf].s[2]  = f2bf(a0.z); qf[kf].s[3]  = f2bf(a0.w);
            qf[kf].s[4]  = f2bf(a1.x); qf[kf].s[5]  = f2bf(a1.y);
            qf[kf].s[6]  = f2bf(a1.z); qf[kf].s[7]  = f2bf(a1.w);
            qf[kf].s[8]  = f2bf(b0.x); qf[kf].s[9]  = f2bf(b0.y);
            qf[kf].s[10] = f2bf(b0.z); qf[kf].s[11] = f2bf(b0.w);
            qf[kf].s[12] = f2bf(b1.x); qf[kf].s[13] = f2bf(b1.y);
            qf[kf].s[14] = f2bf(b1.z); qf[kf].s[15] = f2bf(b1.w);
        }
    }

    v8f o[8];
    #pragma unroll
    for (int t = 0; t < 8; ++t) o[t] = (v8f)0.0f;
    float mrow[8], lrow[8];
    #pragma unroll
    for (int r = 0; r < 8; ++r) { mrow[r] = -1e30f; lrow[r] = 0.0f; }

    const int nchunks = q0 / 32 + 1;              // causal: kv <= q0+15
    for (int j = 0; j < nchunks; ++j) {
        // ---- stage 32 KV rows: K row-major, V transposed, fp32 -> bf16
        {
            int kvr  = threadIdx.x >> 2;           // 0..31
            int part = (threadIdx.x & 3) * 32;     // d offset 0/32/64/96
            const float* kr = qkv + (size_t)(b * Q_LEN + j * 32 + kvr) * QKV_DIM
                              + K_COL0 + hk * HEAD_DIM + part;
            const float* vr = kr + (V_COL0 - K_COL0);
            #pragma unroll
            for (int i = 0; i < 32; i += 4) {
                float4 kq = *(const float4*)(kr + i);
                float4 vq = *(const float4*)(vr + i);
                int d0 = part + i;
                ldsK[kvr * 128 + d0 + 0] = f2bf(kq.x);
                ldsK[kvr * 128 + d0 + 1] = f2bf(kq.y);
                ldsK[kvr * 128 + d0 + 2] = f2bf(kq.z);
                ldsK[kvr * 128 + d0 + 3] = f2bf(kq.w);
                ldsVt[(d0 + 0) * 32 + kvr] = f2bf(vq.x);
                ldsVt[(d0 + 1) * 32 + kvr] = f2bf(vq.y);
                ldsVt[(d0 + 2) * 32 + kvr] = f2bf(vq.z);
                ldsVt[(d0 + 3) * 32 + kvr] = f2bf(vq.w);
            }
        }
        __syncthreads();

        // ---- S = scale * Q K^T  (two 16x16 tiles), causal mask
        float sv[2][8];
        #pragma unroll
        for (int ns = 0; ns < 2; ++ns) {
            v8f sacc = (v8f)0.0f;
            #pragma unroll
            for (int kf = 0; kf < 4; ++kf) {
                Frag kb;                           // B: col = kv (l16), K-dim = d
                int off = (ns * 16 + l16) * 128 + kf * 32 + h16 * 8;
                kb.q[0] = *(const uint4*)&ldsK[off];
                kb.q[1] = *(const uint4*)&ldsK[off + 16];
                sacc = __builtin_amdgcn_wmma_f32_16x16x32_bf16(
                    false, qf[kf].v, false, kb.v, (short)0, sacc, false, false);
            }
            int col  = j * 32 + ns * 16 + l16;
            int rowb = q0 + 8 * h16;
            #pragma unroll
            for (int r = 0; r < 8; ++r) {
                float s = sacc[r] * scale;
                if (col > rowb + r) s = -1e30f;    // causal
                sv[ns][r] = s;
            }
        }

        // ---- online softmax (row = r + 8*h16 lives across 16 lanes)
        float mnew[8], corr[8], pbuf[2][8];
        #pragma unroll
        for (int r = 0; r < 8; ++r) {
            float x = fmaxf(sv[0][r], sv[1][r]);
            #pragma unroll
            for (int off = 8; off > 0; off >>= 1)
                x = fmaxf(x, __shfl_xor(x, off, 16));
            mnew[r] = fmaxf(mrow[r], x);
            corr[r] = __expf(mrow[r] - mnew[r]);
            pbuf[0][r] = __expf(sv[0][r] - mnew[r]);
            pbuf[1][r] = __expf(sv[1][r] - mnew[r]);
            float su = pbuf[0][r] + pbuf[1][r];
            #pragma unroll
            for (int off = 8; off > 0; off >>= 1)
                su += __shfl_xor(su, off, 16);
            lrow[r] = lrow[r] * corr[r] + su;
            mrow[r] = mnew[r];
        }
        #pragma unroll
        for (int t = 0; t < 8; ++t)
            #pragma unroll
            for (int r = 0; r < 8; ++r)
                o[t][r] *= corr[r];

        // ---- P (C layout) -> per-wave LDS -> A-layout fragment
        #pragma unroll
        for (int ns = 0; ns < 2; ++ns)
            #pragma unroll
            for (int r = 0; r < 8; ++r)
                ldsP[w][(r + 8 * h16) * 32 + ns * 16 + l16] = f2bf(pbuf[ns][r]);

        Frag pf;                                   // A: row = l16, kv chunks
        {
            int off = l16 * 32 + h16 * 8;
            pf.q[0] = *(const uint4*)&ldsP[w][off];
            pf.q[1] = *(const uint4*)&ldsP[w][off + 16];
        }
        // ---- O += P V   (8 d-tiles, B from transposed V)
        #pragma unroll
        for (int t = 0; t < 8; ++t) {
            Frag vf;
            int off = (t * 16 + l16) * 32 + h16 * 8;
            vf.q[0] = *(const uint4*)&ldsVt[off];
            vf.q[1] = *(const uint4*)&ldsVt[off + 16];
            o[t] = __builtin_amdgcn_wmma_f32_16x16x32_bf16(
                false, pf.v, false, vf.v, (short)0, o[t], false, false);
        }
        __syncthreads();
    }

    // ---- normalize and store fp32 attention output
    #pragma unroll
    for (int t = 0; t < 8; ++t) {
        float* op = attn + (size_t)(b * Q_LEN + q0 + 8 * h16) * Q_COLS
                    + h * HEAD_DIM + t * 16 + l16;
        #pragma unroll
        for (int r = 0; r < 8; ++r)
            op[(size_t)r * Q_COLS] = o[t][r] * (1.0f / lrow[r]);
    }
}

// ---------------------------------------------------------------------------
// Launch. Inputs (setup_inputs order):
//  0 hidden_states f32 [4096,4096]   1 Wqkv_w f32 [6144,4096]
//  2 out_proj_w f32 [4096,4096]      3 inv_freq f32 [64]
//  4 k_cache, 5 v_cache, 6 block_offsets  -> identity round-trip, unused
//  7 position_ids i32 [4096]
// Workspace: qkv f32 (96MB) | attn f32 (64MB)
// ---------------------------------------------------------------------------
extern "C" void kernel_launch(void* const* d_in, const int* in_sizes, int n_in,
                              void* d_out, int out_size, void* d_ws, size_t ws_size,
                              hipStream_t stream) {
    const float* hidden   = (const float*)d_in[0];
    const float* wqkv     = (const float*)d_in[1];
    const float* wout     = (const float*)d_in[2];
    const float* inv_freq = (const float*)d_in[3];
    const int*   pos_ids  = (const int*)  d_in[7];
    float* out  = (float*)d_out;
    float* qkv  = (float*)d_ws;
    float* attn = qkv + (size_t)TOTAL * QKV_DIM;

    // 1) qkv = hidden @ Wqkv^T          (M=4096, N=6144, K=4096)
    gemm_bf16_tn<<<dim3(QKV_DIM / 128, TOTAL / 128), 256, 0, stream>>>(
        hidden, wqkv, qkv, TOTAL, QKV_DIM, D_MODEL);

    // 2) clamp(+-8) + RoPE in place
    {
        int total = TOTAL * ROPE_HEADS * (HEAD_DIM / 2);
        rope_clamp_kernel<<<(total + 255) / 256, 256, 0, stream>>>(qkv, inv_freq, pos_ids);
    }

    // 3) causal GQA flash attention
    attn_kernel<<<dim3(Q_LEN / 16, N_KV, BATCH), 128, 0, stream>>>(qkv, attn);

    // 4) out = attn @ out_proj^T        (M=4096, N=4096, K=4096)
    gemm_bf16_tn<<<dim3(D_MODEL / 128, TOTAL / 128), 256, 0, stream>>>(
        attn, wout, out, TOTAL, D_MODEL, Q_COLS);
}
